// SplineConv_8512625180753
// MI455X (gfx1250) — matrix operand order repaired and checked by reading, not compiled, not run on hardware
//
#include <hip/hip_runtime.h>

typedef __attribute__((ext_vector_type(16))) _Float16 v16h;
typedef __attribute__((ext_vector_type(8)))  _Float16 v8h;
typedef __attribute__((ext_vector_type(8)))  float    v8f;

#define SP_ORDER   2
#define SP_H       7
#define SP_NKNOT   10
#define SP_DIM     1024      /* OUT*IN */
#define SP_KPAD    64        /* 49 grid cells padded to 64 */
#define PTS_PER_WG 128
#define THREADS    256
#define CH_CHUNK   256
#define N_CHUNKS   (SP_DIM / CH_CHUNK)

__device__ __forceinline__ float sp_sdiv(float n, float d) {
    return n / (d == 0.0f ? 1.0f : d);
}

// Quadratic B-spline basis values (traces the reference de Boor alphas).
__device__ __forceinline__ void sp_basis3(const float* __restrict__ T, float x,
                                          int k, float* B) {
    float a10 = sp_sdiv(x - T[k - 1], T[k + 1] - T[k - 1]);
    float a11 = sp_sdiv(x - T[k],     T[k + 2] - T[k]);
    float a21 = sp_sdiv(x - T[k],     T[k + 1] - T[k]);
    B[0] = (1.0f - a21) * (1.0f - a10);
    B[1] = (1.0f - a21) * a10 + a21 * (1.0f - a11);
    B[2] = a21 * a11;
}

__global__ __launch_bounds__(THREADS)
void spline_wmma_kernel(const float* __restrict__ xy,
                        const float* __restrict__ Tx,
                        const float* __restrict__ Ty,
                        const float* __restrict__ C,
                        float* __restrict__ out) {
    __shared__ __attribute__((aligned(32))) _Float16 Wlds[PTS_PER_WG * SP_KPAD]; // 16 KB
    __shared__ __attribute__((aligned(32))) _Float16 Clds[CH_CHUNK * SP_KPAD];   // 32 KB

    const int t    = threadIdx.x;
    const int lane = t & 31;
    const int wv   = t >> 5;          // wave id (wave32)
    const int col  = lane & 15;
    const int g    = lane >> 4;       // lane half-group
    const int n0   = blockIdx.x * PTS_PER_WG;

    // ---- Phase 1: per-point densified 1x64 f16 weight row ----
    if (t < PTS_PER_WG) {
        const int n = n0 + t;
        const float x = xy[2 * n + 0];
        const float y = xy[2 * n + 1];
        int ssx = 0, ssy = 0;
        #pragma unroll
        for (int i = 0; i < SP_NKNOT; ++i) {
            ssx += (Tx[i] <= x) ? 1 : 0;   // searchsorted(side='right')
            ssy += (Ty[i] <= y) ? 1 : 0;
        }
        int kx = (ssx == SP_NKNOT) ? (SP_NKNOT - SP_ORDER - 2) : (ssx - 1);
        int ky = (ssy == SP_NKNOT) ? (SP_NKNOT - SP_ORDER - 2) : (ssy - 1);
        float Bx[3], By[3];
        sp_basis3(Tx, x, kx, Bx);
        sp_basis3(Ty, y, ky, By);
        _Float16* wr = &Wlds[t * SP_KPAD];
        #pragma unroll
        for (int gg = 0; gg < SP_KPAD; ++gg) wr[gg] = (_Float16)0.0f;
        #pragma unroll
        for (int a = 0; a < 3; ++a)
            #pragma unroll
            for (int b = 0; b < 3; ++b)
                wr[(kx - 2 + a) * SP_H + (ky - 2 + b)] = (_Float16)(Bx[a] * By[b]);
    }
    __syncthreads();

    // ---- Phase 2: load A fragments (16 points x K=64, ISA 16-bit A layout) ----
    // lane L: M = L%16, g = L/16; VGPR v halves at K-offsets {8g..8g+7} U {16+8g..}
    const _Float16* arow = &Wlds[(wv * 16 + col) * SP_KPAD];
    union { v16h v; v8h h[2]; } a0, a1;
    a0.h[0] = *(const v8h*)(arow +  0 + 8 * g);
    a0.h[1] = *(const v8h*)(arow + 16 + 8 * g);
    a1.h[0] = *(const v8h*)(arow + 32 + 8 * g);
    a1.h[1] = *(const v8h*)(arow + 48 + 8 * g);

    for (int chunk = 0; chunk < N_CHUNKS; ++chunk) {
        __syncthreads();   // previous chunk's B reads complete before restage
        // stage 256 channels as f16, K-major per channel: Clds[d][K]
        {
            const int d = chunk * CH_CHUNK + t;
            const float* src = C + (size_t)d * (SP_H * SP_H);
            _Float16* dst = &Clds[t * SP_KPAD];
            #pragma unroll
            for (int gg = 0; gg < SP_H * SP_H; ++gg) dst[gg] = (_Float16)src[gg];
            #pragma unroll
            for (int gg = SP_H * SP_H; gg < SP_KPAD; ++gg) dst[gg] = (_Float16)0.0f;
        }
        __syncthreads();

        for (int ct = 0; ct < CH_CHUNK / 16; ++ct) {
            // B fragment, lane L: N = L%16, K = {8g..8g+7} U {16+8g..16+8g+7}
            const _Float16* brow = &Clds[(ct * 16 + col) * SP_KPAD];
            v16h b0 = __builtin_shufflevector(*(const v8h*)(brow +  0 + 8 * g), *(const v8h*)(brow + 16 + 8 * g), 0,1,2,3,4,5,6,7,8,9,10,11,12,13,14,15);
            v16h b1 = __builtin_shufflevector(*(const v8h*)(brow + 32 + 8 * g), *(const v8h*)(brow + 48 + 8 * g), 0,1,2,3,4,5,6,7,8,9,10,11,12,13,14,15);
            v8f acc = {};
            acc = __builtin_amdgcn_wmma_f32_16x16x32_f16(false, a0.v, false, b0,
                                                         (short)0, acc, false, false);
            acc = __builtin_amdgcn_wmma_f32_16x16x32_f16(false, a1.v, false, b1,
                                                         (short)0, acc, false, false);
            // D layout: lane col = L%16, VGPR r -> row M = r + 8g
            float* obase = out + (size_t)(n0 + wv * 16 + 8 * g) * SP_DIM
                               + chunk * CH_CHUNK + ct * 16 + col;
            #pragma unroll
            for (int r = 0; r < 8; ++r)
                obase[(size_t)r * SP_DIM] = acc[r];
        }
    }
}

extern "C" void kernel_launch(void* const* d_in, const int* in_sizes, int n_in,
                              void* d_out, int out_size, void* d_ws, size_t ws_size,
                              hipStream_t stream) {
    (void)n_in; (void)out_size; (void)d_ws; (void)ws_size;
    const float* xy = (const float*)d_in[0];
    const float* Tx = (const float*)d_in[1];
    const float* Ty = (const float*)d_in[2];
    const float* C  = (const float*)d_in[3];
    float* out = (float*)d_out;
    const int N = in_sizes[0] / 2;          // 32768 points
    const int grid = N / PTS_PER_WG;        // 256 workgroups
    spline_wmma_kernel<<<grid, THREADS, 0, stream>>>(xy, Tx, Ty, C, out);
}
